// MultiheadAttentionWithRelativePE_79061757985075
// MI455X (gfx1250) — matrix operand order repaired and checked by reading, not compiled, not run on hardware
//
#include <hip/hip_runtime.h>

// ---------------------------------------------------------------------------
// MultiheadAttentionWithRelativePE for MI455X (gfx1250, wave32, WMMA)
//   B=2, N=2048, C=1024, H=16, Dh=64
// Pipeline:
//   0) fp32 -> bf16 conversion of x and the four weight matrices
//   1) Q/K/V projections: WMMA bf16 GEMM (Y = X @ W^T + b), bf16 output
//   2) V transpose to [b,h,d,j] so PV's B-operand has contiguous K
//   3) FUSED: scores = QK^T/8 + bias -> softmax -> attn_weights (NT store,
//      mandatory output) -> O = P@V, all in one kernel (P staged in LDS)
//   4) out = O @ Wo^T + bo (WMMA, fp32 out)
// Bandwidth: bias read (512MB, NT) + attn write (512MB, NT) dominate; all
// weights/Q/K/V (~50MB bf16) are L2-resident (192MB L2).
// ---------------------------------------------------------------------------

typedef __bf16 bf16_t;
typedef __attribute__((ext_vector_type(16))) __bf16 v16bf;
typedef __attribute__((ext_vector_type(8)))  __bf16 v8bf;
typedef __attribute__((ext_vector_type(8)))  float  v8f;

#define BATCH 2
#define SEQ 2048
#define DMODEL 1024
#define NHEADS 16
#define HDIM 64

// ---------------- helpers ----------------

__device__ __forceinline__ bf16_t to_bf16(float f) {
  unsigned u = __builtin_bit_cast(unsigned, f);
  unsigned r = u + 0x7FFFu + ((u >> 16) & 1u);   // round-to-nearest-even
  unsigned short h = (unsigned short)(r >> 16);
  return __builtin_bit_cast(bf16_t, h);
}

// A-fragment (16x32 bf16): lane L&15 holds row L&15; K chunks {k0..k0+7} and
// {k0+16..k0+23} with k0 = kk + (L>=16 ? 8 : 0). Two contiguous 16B loads.
__device__ __forceinline__ v16bf load_a_bf16(const bf16_t* __restrict__ A,
                                             long lda, int row, int kk, int lane) {
  int r  = row + (lane & 15);
  int k0 = kk + ((lane >> 4) << 3);
  const bf16_t* p = A + (long)r * lda + k0;
  union { v16bf v; v8bf h[2]; } u;
  u.h[0] = *(const v8bf*)(p);
  u.h[1] = *(const v8bf*)(p + 16);
  return u.v;
}

// B-fragment (32x16 bf16) from "W[n,k]" orientation: lane L holds col L&15,
// K chunks {k0..k0+7} and {k0+16..k0+23} with k0 = kk + (L>=16 ? 8 : 0).
__device__ __forceinline__ v16bf load_b_bf16(const bf16_t* __restrict__ W,
                                             long ldw, int col, int kk, int lane) {
  int n  = col + (lane & 15);
  int k0 = kk + ((lane >> 4) << 3);
  const bf16_t* p = W + (long)n * ldw + k0;
  union { v16bf v; v8bf h[2]; } u;
  u.h[0] = *(const v8bf*)(p);
  u.h[1] = *(const v8bf*)(p + 16);
  return u.v;
}

#define WMMA_BF16(a, b, c) \
  __builtin_amdgcn_wmma_f32_16x16x32_bf16(false, (a), false, (b), (short)0, (c), false, false)

// ---------------- kernel 0: fp32 -> bf16 ----------------

__global__ void cvt_f32_bf16_kernel(const float* __restrict__ src,
                                    bf16_t* __restrict__ dst, int n) {
  int i = blockIdx.x * blockDim.x + threadIdx.x;
  if (i < n) dst[i] = to_bf16(src[i]);
}

// ---------------- kernels 1/4: Y = X @ W^T + bias ----------------
// A [M,K] bf16 row-major, W [N,K] bf16 row-major. Compile-time N,K and output
// type (no branches in epilogue; stores use immediate offsets).
// Block tile 128x128, 8 waves (4 row x 2 col), wave tile 32x64.

template <int N, int K, bool STORE_BF16>
__global__ __launch_bounds__(256)
void gemm_xw_kernel(const bf16_t* __restrict__ A, const bf16_t* __restrict__ W,
                    const float* __restrict__ bias, void* __restrict__ out) {
  const int lane = threadIdx.x & 31;
  const int wave = threadIdx.x >> 5;
  const int wr = wave & 3, wc = wave >> 2;
  const int m0 = blockIdx.x * 128 + wr * 32;
  const int n0 = blockIdx.y * 128 + wc * 64;

  v8f acc[2][4];
#pragma unroll
  for (int i = 0; i < 2; ++i)
#pragma unroll
    for (int j = 0; j < 4; ++j) acc[i][j] = (v8f)0.0f;

  for (int kk = 0; kk < K; kk += 32) {
    v16bf a[2], b[4];
#pragma unroll
    for (int i = 0; i < 2; ++i) a[i] = load_a_bf16(A, K, m0 + 16 * i, kk, lane);
#pragma unroll
    for (int j = 0; j < 4; ++j) b[j] = load_b_bf16(W, K, n0 + 16 * j, kk, lane);
#pragma unroll
    for (int i = 0; i < 2; ++i)
#pragma unroll
      for (int j = 0; j < 4; ++j) acc[i][j] = WMMA_BF16(a[i], b[j], acc[i][j]);
  }

  // D layout: elem e <-> row (e + (lane>>4)*8), lane&15 <-> col
#pragma unroll
  for (int i = 0; i < 2; ++i) {
#pragma unroll
    for (int j = 0; j < 4; ++j) {
      int col = n0 + 16 * j + (lane & 15);
      float bv = bias[col];
      int rowb = m0 + 16 * i + ((lane >> 4) << 3);
#pragma unroll
      for (int e = 0; e < 8; ++e) {
        float v = acc[i][j][e] + bv;
        long idx = (long)(rowb + e) * N + col;
        if (STORE_BF16) ((bf16_t*)out)[idx] = to_bf16(v);
        else            ((float*)out)[idx]  = v;
      }
    }
  }
}

// ---------------- kernel 2: V transpose to [b,h,d,j] ----------------

__global__ void transpose_v_kernel(const bf16_t* __restrict__ Vp,
                                   bf16_t* __restrict__ Vt) {
  long idx = (long)blockIdx.x * blockDim.x + threadIdx.x;  // [b,h,d,j] flat
  int j = idx & (SEQ - 1);
  long r = idx >> 11;
  int d = r & (HDIM - 1); r >>= 6;
  int h = r & (NHEADS - 1);
  int b = (int)(r >> 4);
  Vt[idx] = Vp[((long)(b * SEQ + j)) * DMODEL + h * HDIM + d];
}

// ---------------- kernel 3: fused scores+bias+softmax+PV ----------------
// One workgroup = 16 rows x 2048 cols of one (b,h). 8 waves: wave w owns the
// 16x256 column strip [w*256, (w+1)*256).
//   scores:  16 WMMA tiles/wave x 2 k-steps (K=64)
//   softmax: shfl_xor within 16-lane row groups + LDS cross-wave reduction
//   P out:   fp32 nontemporal stores (attn_weights output), bf16 copy in LDS
//   PV:      wave w reduces K-strip [w*256,(w+1)*256): 8 k-steps x 4 tiles,
//            partial O accumulated across waves with ds_add_f32 (shared atomics)

__global__ __launch_bounds__(256)
void attn_fused_kernel(const bf16_t* __restrict__ Q, const bf16_t* __restrict__ Km,
                       const bf16_t* __restrict__ Vt, const float* __restrict__ bias,
                       float* __restrict__ attnW, bf16_t* __restrict__ Oh) {
  // P row stride padded by 8 bf16 (16B) => row r starts at bank 4r: the 16
  // b128 A-fragment reads (one per row) hit disjoint bank groups.
  __shared__ __align__(16) bf16_t Plds[16][SEQ + 8];
  __shared__ float Olds[16][HDIM];
  __shared__ float redmax[8][16];
  __shared__ float redsum[8][16];

  const int bh = blockIdx.z;               // b*16 + h
  const int b = bh >> 4, h = bh & 15;
  const int lane = threadIdx.x & 31;
  const int wave = threadIdx.x >> 5;
  const int m0 = blockIdx.x * 16;          // row block within SEQ

  const bf16_t* Qb = Q  + ((long)b * SEQ) * DMODEL + h * HDIM;
  const bf16_t* Kb = Km + ((long)b * SEQ) * DMODEL + h * HDIM;
  const bf16_t* Vb = Vt + (long)bh * HDIM * SEQ;
  const float*  Bb = bias  + (long)bh * SEQ * SEQ;
  float*        Sb = attnW + (long)bh * SEQ * SEQ;

  // zero O accumulator (covered by the first __syncthreads below)
  for (int i = threadIdx.x; i < 16 * HDIM; i += 256) ((float*)Olds)[i] = 0.0f;

  // ---- scores: 16 tiles of 16x16 per wave over K=64 ----
  const int n0 = wave * 256;
  v8f acc[16];
#pragma unroll
  for (int t = 0; t < 16; ++t) acc[t] = (v8f)0.0f;

#pragma unroll
  for (int kk = 0; kk < HDIM; kk += 32) {
    v16bf a = load_a_bf16(Qb, DMODEL, m0, kk, lane);
#pragma unroll
    for (int t = 0; t < 16; ++t) {
      v16bf bf = load_b_bf16(Kb, DMODEL, n0 + 16 * t, kk, lane);
      acc[t] = WMMA_BF16(a, bf, acc[t]);
    }
  }

  // ---- scale + relative bias (streaming NT reads) ----
  const int colbase = n0 + (lane & 15);
  const int lr0 = (lane >> 4) << 3;        // local row base (0 or 8)
  const int rowb = m0 + lr0;               // global row base
#pragma unroll
  for (int t = 0; t < 16; ++t) {
    const float* bp = Bb + (long)rowb * SEQ + colbase + 16 * t;
#pragma unroll
    for (int e = 0; e < 8; ++e)
      acc[t][e] = acc[t][e] * 0.125f + __builtin_nontemporal_load(bp + (long)e * SEQ);
  }

  // ---- row max: in-lane over tiles, shfl within 16-lane group, LDS x-wave ----
  float rm[8];
#pragma unroll
  for (int e = 0; e < 8; ++e) {
    float m = acc[0][e];
#pragma unroll
    for (int t = 1; t < 16; ++t) m = fmaxf(m, acc[t][e]);
#pragma unroll
    for (int off = 8; off > 0; off >>= 1) m = fmaxf(m, __shfl_xor(m, off, 32));
    rm[e] = m;
  }
  if ((lane & 15) == 0) {
#pragma unroll
    for (int e = 0; e < 8; ++e) redmax[wave][lr0 + e] = rm[e];
  }
  __syncthreads();
  float fm[8];
#pragma unroll
  for (int e = 0; e < 8; ++e) {
    float m = redmax[0][lr0 + e];
#pragma unroll
    for (int w = 1; w < 8; ++w) m = fmaxf(m, redmax[w][lr0 + e]);
    fm[e] = m;
  }

  // ---- exp + row sum ----
  float ls[8];
#pragma unroll
  for (int e = 0; e < 8; ++e) ls[e] = 0.0f;
#pragma unroll
  for (int t = 0; t < 16; ++t) {
#pragma unroll
    for (int e = 0; e < 8; ++e) {
      float p = __expf(acc[t][e] - fm[e]);
      acc[t][e] = p;
      ls[e] += p;
    }
  }
#pragma unroll
  for (int e = 0; e < 8; ++e) {
#pragma unroll
    for (int off = 8; off > 0; off >>= 1) ls[e] += __shfl_xor(ls[e], off, 32);
  }
  if ((lane & 15) == 0) {
#pragma unroll
    for (int e = 0; e < 8; ++e) redsum[wave][lr0 + e] = ls[e];
  }
  __syncthreads();
  float inv[8];
#pragma unroll
  for (int e = 0; e < 8; ++e) {
    float s = redsum[0][lr0 + e];
#pragma unroll
    for (int w = 1; w < 8; ++w) s += redsum[w][lr0 + e];
    inv[e] = 1.0f / s;
  }

  // ---- normalized P: NT store to attn_weights output + bf16 copy into LDS ----
#pragma unroll
  for (int t = 0; t < 16; ++t) {
    float* sp = Sb + (long)rowb * SEQ + colbase + 16 * t;
#pragma unroll
    for (int e = 0; e < 8; ++e) {
      float p = acc[t][e] * inv[e];
      __builtin_nontemporal_store(p, sp + (long)e * SEQ);
      Plds[lr0 + e][n0 + 16 * t + (lane & 15)] = to_bf16(p);
    }
  }
  __syncthreads();

  // ---- O = P @ V over this wave's K strip ----
  v8f oacc[4];
#pragma unroll
  for (int j = 0; j < 4; ++j) oacc[j] = (v8f)0.0f;

#pragma unroll
  for (int ks = 0; ks < 8; ++ks) {
    int kk = n0 + ks * 32;
    // A-fragment from LDS P: row = lane&15, K chunks k0 and k0+16
    int k0 = kk + ((lane >> 4) << 3);
    const bf16_t* pp = &Plds[lane & 15][k0];
    union { v16bf v; v8bf hh[2]; } ua;
    ua.hh[0] = *(const v8bf*)(pp);
    ua.hh[1] = *(const v8bf*)(pp + 16);
#pragma unroll
    for (int j = 0; j < 4; ++j) {
      v16bf bf = load_b_bf16(Vb, SEQ, 16 * j, kk, lane);
      oacc[j] = WMMA_BF16(ua.v, bf, oacc[j]);
    }
  }

  // cross-wave O reduction in LDS
#pragma unroll
  for (int j = 0; j < 4; ++j) {
    int d = 16 * j + (lane & 15);
#pragma unroll
    for (int e = 0; e < 8; ++e) atomicAdd(&Olds[lr0 + e][d], oacc[j][e]);
  }
  __syncthreads();

  // store O block (bf16, head-interleaved [B*N, C])
  for (int i = threadIdx.x; i < 16 * HDIM; i += 256) {
    int r = i >> 6, d = i & 63;
    Oh[((long)(b * SEQ + m0 + r)) * DMODEL + h * HDIM + d] = to_bf16(Olds[r][d]);
  }
}

// ---------------- host ----------------

extern "C" void kernel_launch(void* const* d_in, const int* in_sizes, int n_in,
                              void* d_out, int out_size, void* d_ws, size_t ws_size,
                              hipStream_t stream) {
  const float* x  = (const float*)d_in[0];
  const float* rb = (const float*)d_in[1];
  const float* wq = (const float*)d_in[2];
  const float* bq = (const float*)d_in[3];
  const float* wk = (const float*)d_in[4];
  const float* bk = (const float*)d_in[5];
  const float* wv = (const float*)d_in[6];
  const float* bv = (const float*)d_in[7];
  const float* wo = (const float*)d_in[8];
  const float* bo = (const float*)d_in[9];

  float* out   = (float*)d_out;
  float* attnW = out + (size_t)BATCH * SEQ * DMODEL;  // [B,H,N,N] fp32 output

  char* ws = (char*)d_ws;
  const size_t MB = 1024 * 1024;
  bf16_t* xh  = (bf16_t*)(ws + 0 * MB);   // [4096,1024] bf16  (8 MB)
  bf16_t* wqh = (bf16_t*)(ws + 8 * MB);   // 2 MB each
  bf16_t* wkh = (bf16_t*)(ws + 10 * MB);
  bf16_t* wvh = (bf16_t*)(ws + 12 * MB);
  bf16_t* woh = (bf16_t*)(ws + 14 * MB);
  bf16_t* qh  = (bf16_t*)(ws + 16 * MB);  // [4096,1024] bf16
  bf16_t* kh  = (bf16_t*)(ws + 24 * MB);
  bf16_t* vh  = (bf16_t*)(ws + 32 * MB);
  bf16_t* vt  = (bf16_t*)(ws + 40 * MB);  // [B,H,Dh,N] bf16
  bf16_t* oh  = (bf16_t*)(ws + 48 * MB);  // [4096,1024] bf16

  const int NX = BATCH * SEQ * DMODEL;    // 4194304
  const int NW = DMODEL * DMODEL;         // 1048576

  // 0) conversions
  cvt_f32_bf16_kernel<<<NX / 256, 256, 0, stream>>>(x,  xh,  NX);
  cvt_f32_bf16_kernel<<<NW / 256, 256, 0, stream>>>(wq, wqh, NW);
  cvt_f32_bf16_kernel<<<NW / 256, 256, 0, stream>>>(wk, wkh, NW);
  cvt_f32_bf16_kernel<<<NW / 256, 256, 0, stream>>>(wv, wvh, NW);
  cvt_f32_bf16_kernel<<<NW / 256, 256, 0, stream>>>(wo, woh, NW);

  // 1) projections (M=4096, N=1024, K=1024), bf16 output
  dim3 gproj(BATCH * SEQ / 128, DMODEL / 128);
  gemm_xw_kernel<DMODEL, DMODEL, true><<<gproj, 256, 0, stream>>>(xh, wqh, bq, qh);
  gemm_xw_kernel<DMODEL, DMODEL, true><<<gproj, 256, 0, stream>>>(xh, wkh, bk, kh);
  gemm_xw_kernel<DMODEL, DMODEL, true><<<gproj, 256, 0, stream>>>(xh, wvh, bv, vh);

  // 2) V transpose
  transpose_v_kernel<<<NX / 256, 256, 0, stream>>>(vh, vt);

  // 3) fused scores + bias + softmax + attn-weight output + PV
  dim3 gf(SEQ / 16, 1, BATCH * NHEADS);
  attn_fused_kernel<<<gf, 256, 0, stream>>>(qh, kh, vt, rb, attnW, oh);

  // 4) out = O @ Wo^T + bo (fp32)
  gemm_xw_kernel<DMODEL, DMODEL, false><<<gproj, 256, 0, stream>>>(oh, woh, bo, out);
}
